// MultiHeadAttention_9904194584650
// MI455X (gfx1250) — matrix operand order repaired and checked by reading, not compiled, not run on hardware
//
#include <hip/hip_runtime.h>

// ---------------------------------------------------------------------------
// MHA forward for MI455X (gfx1250, wave32, WMMA f16->f32).
//   B=4, S=2048, D=1024, H=16, HD=64.
// Three kernels: QKV GEMM -> flash attention -> output GEMM.
// Workspace: f16 Q,K,V in [B,H,S,HD] + f16 O in [B,S,H,HD]  (4 x 16 MB).
// ---------------------------------------------------------------------------

typedef __attribute__((ext_vector_type(16))) _Float16 v16h;
typedef __attribute__((ext_vector_type(8)))  float    v8f;

union Frag16 { v16h h; unsigned int u[8]; };

#define BATCH    4
#define SEQ      2048
#define DMODEL   1024
#define NHEADS   16
#define HEADDIM  64
#define MTOT     (BATCH * SEQ)          // 8192 rows of tokens

// A-matrix (16x32 f16) K index for VGPR v (2 halfs per VGPR), lane-half hf.
__device__ __forceinline__ int a_kidx(int v, int hf) {
    return ((v & 4) ? 16 : 0) + 2 * (v & 3) + 8 * hf;
}
// B-matrix (32x16 f16) K index for VGPR v, lane-half hf.
__device__ __forceinline__ int b_kidx(int v, int hf) {
    return ((v & 4) ? 16 : 0) + 2 * (v & 3) + 8 * hf;
}

__device__ __forceinline__ v8f wmma_f16(const Frag16& a, const Frag16& b, v8f c) {
    return __builtin_amdgcn_wmma_f32_16x16x32_f16(
        false, a.h, false, b.h, (short)0, c, false, false);
}

// ---------------------------------------------------------------------------
// GEMM: C[8192 x N] = A[8192 x K] * B[K x N] + bias.
// MODE 0: A = x (f32), B = w_qkv, N = 3072. Output scattered to f16 Q/K/V
//         in [B,H,S,HD] layout.
// MODE 1: A = O (f16),  B = w_out, N = 1024. Output f32 to d_out + bias.
// Block: 256 threads (8 waves), tile 128x128, K-step 32.
// Wave (wm = wave&3, wn = wave>>2) owns a 32x64 sub-tile = 2x4 WMMA tiles.
// ---------------------------------------------------------------------------
#define LDSTRIDE 40   // 32 + 8 pad halfs; 80 bytes = multiple of 16

template <int MODE>
__global__ __launch_bounds__(256) void gemm128(
    const float* __restrict__ Af32, const _Float16* __restrict__ Af16,
    const float* __restrict__ Bm,   const float* __restrict__ bias,
    _Float16* __restrict__ qf, _Float16* __restrict__ kf, _Float16* __restrict__ vf,
    float* __restrict__ out, int K, int N)
{
    const int tid  = threadIdx.x;
    const int lane = tid & 31;
    const int wave = tid >> 5;
    const int lo   = lane & 15;
    const int hf   = lane >> 4;
    const int bM   = blockIdx.y * 128;
    const int bN   = blockIdx.x * 128;
    const int wm   = wave & 3;
    const int wn   = wave >> 2;

    __shared__ _Float16 As[128 * LDSTRIDE];   // [row][k]  (row-major in k)
    __shared__ _Float16 Bt[128 * LDSTRIDE];   // [n][k]    (transposed)

    v8f acc[2][4];
#pragma unroll
    for (int mi = 0; mi < 2; ++mi)
#pragma unroll
        for (int ni = 0; ni < 4; ++ni)
            acc[mi][ni] = (v8f){0.f, 0.f, 0.f, 0.f, 0.f, 0.f, 0.f, 0.f};

    for (int k0 = 0; k0 < K; k0 += 32) {
        __syncthreads();
        // ---- stage A tile (128 x 32) into LDS as f16 -----------------------
        if (MODE == 0) {
#pragma unroll
            for (int i = 0; i < 4; ++i) {
                int idx = tid + i * 256;              // 1024 float4 slots
                int row = idx >> 3, c4 = (idx & 7) * 4;
                float4 f = *(const float4*)(Af32 + (size_t)(bM + row) * K + k0 + c4);
                _Float16* dst = &As[row * LDSTRIDE + c4];
                dst[0] = (_Float16)f.x; dst[1] = (_Float16)f.y;
                dst[2] = (_Float16)f.z; dst[3] = (_Float16)f.w;
            }
        } else {
#pragma unroll
            for (int i = 0; i < 2; ++i) {
                int idx = tid + i * 256;              // 512 uint4 slots
                int row = idx >> 2, c8 = (idx & 3) * 8;
                uint4 u = *(const uint4*)(Af16 + (size_t)(bM + row) * K + k0 + c8);
                *(uint4*)&As[row * LDSTRIDE + c8] = u;
            }
        }
        // ---- stage B tile (32 x 128) into LDS transposed as f16 ------------
#pragma unroll
        for (int i = 0; i < 4; ++i) {
            int idx = tid + i * 256;                  // 1024 float4 slots
            int r = idx >> 5, c4 = (idx & 31) * 4;
            float4 f = *(const float4*)(Bm + (size_t)(k0 + r) * N + bN + c4);
            Bt[(c4 + 0) * LDSTRIDE + r] = (_Float16)f.x;
            Bt[(c4 + 1) * LDSTRIDE + r] = (_Float16)f.y;
            Bt[(c4 + 2) * LDSTRIDE + r] = (_Float16)f.z;
            Bt[(c4 + 3) * LDSTRIDE + r] = (_Float16)f.w;
        }
        __syncthreads();

        // ---- fragments + 8 WMMA ------------------------------------------
        Frag16 a[2], b[4];
#pragma unroll
        for (int mi = 0; mi < 2; ++mi) {
            int rA = wm * 32 + mi * 16 + lo;
#pragma unroll
            for (int v = 0; v < 8; ++v)
                a[mi].u[v] = *(const unsigned int*)&As[rA * LDSTRIDE + a_kidx(v, hf)];
        }
#pragma unroll
        for (int ni = 0; ni < 4; ++ni) {
            int cB = wn * 64 + ni * 16 + lo;
#pragma unroll
            for (int v = 0; v < 8; ++v)
                b[ni].u[v] = *(const unsigned int*)&Bt[cB * LDSTRIDE + b_kidx(v, hf)];
        }
#pragma unroll
        for (int mi = 0; mi < 2; ++mi)
#pragma unroll
            for (int ni = 0; ni < 4; ++ni)
                acc[mi][ni] = wmma_f16(a[mi], b[ni], acc[mi][ni]);
    }

    // ---- epilogue ---------------------------------------------------------
#pragma unroll
    for (int mi = 0; mi < 2; ++mi)
#pragma unroll
        for (int ni = 0; ni < 4; ++ni)
#pragma unroll
            for (int r = 0; r < 8; ++r) {
                int gm = bM + wm * 32 + mi * 16 + r + hf * 8;
                int gn = bN + wn * 64 + ni * 16 + lo;
                float val = acc[mi][ni][r] + bias[gn];
                if (MODE == 0) {
                    int which = gn >> 10;          // 0=Q 1=K 2=V
                    int d  = gn & 1023;
                    int h  = d >> 6, hd = d & 63;
                    int bb = gm >> 11, s = gm & 2047;
                    _Float16* dst = (which == 0) ? qf : (which == 1 ? kf : vf);
                    dst[(((size_t)(bb * NHEADS + h) * SEQ + s) << 6) + hd] = (_Float16)val;
                } else {
                    out[(size_t)gm * N + gn] = val;
                }
            }
}

// ---------------------------------------------------------------------------
// Flash attention, causal. Block = 128 threads (4 waves) owns a 64-query tile
// of one (batch, head). Each wave owns 16 queries. K tiles of 64 keys.
// Never materializes the score matrix in HBM.
// ---------------------------------------------------------------------------
#define KVSTRIDE 72   // 64 + 8 pad halfs; 144 bytes = multiple of 16

__global__ __launch_bounds__(128) void flash_attn(
    const _Float16* __restrict__ qf, const _Float16* __restrict__ kf,
    const _Float16* __restrict__ vf, _Float16* __restrict__ obuf)
{
    const int tid  = threadIdx.x;
    const int lane = tid & 31;
    const int wave = tid >> 5;
    const int lo   = lane & 15;
    const int hf   = lane >> 4;
    const int qt   = blockIdx.x;           // query tile
    const int bh   = blockIdx.y;           // batch*NHEADS + head
    const int qbase = qt * 64;

    const _Float16* Q  = qf + (size_t)bh * SEQ * HEADDIM;
    const _Float16* Kp = kf + (size_t)bh * SEQ * HEADDIM;
    const _Float16* Vp = vf + (size_t)bh * SEQ * HEADDIM;

    __shared__ _Float16 Klds[64 * KVSTRIDE];        // [key][hd]   row-major
    __shared__ _Float16 Vt[64 * KVSTRIDE];          // [hd][key]   transposed
    __shared__ _Float16 Plds[4 * 16 * KVSTRIDE];    // per-wave P strip [m][key]

    // Q fragments (16 queries x 64 hd = 2 K-chunks), kept in registers.
    Frag16 qa[2];
    {
        int qrow = qbase + wave * 16 + lo;
#pragma unroll
        for (int kc = 0; kc < 2; ++kc)
#pragma unroll
            for (int v = 0; v < 8; ++v) {
                int kk = kc * 32 + a_kidx(v, hf);
                qa[kc].u[v] = *(const unsigned int*)(Q + (size_t)qrow * HEADDIM + kk);
            }
    }

    v8f oacc[4];
#pragma unroll
    for (int ni = 0; ni < 4; ++ni)
        oacc[ni] = (v8f){0.f, 0.f, 0.f, 0.f, 0.f, 0.f, 0.f, 0.f};
    float mprev[8], lsum[8];
#pragma unroll
    for (int r = 0; r < 8; ++r) { mprev[r] = -3.0e38f; lsum[r] = 0.f; }

    const float scale = 0.125f;  // 1/sqrt(64)

    for (int j = 0; j <= qt; ++j) {         // causal: skip tiles above diagonal
        const int kb = j * 64;
        __syncthreads();
        // ---- stage K tile (row-major) and V tile (transposed) --------------
#pragma unroll
        for (int i = 0; i < 4; ++i) {
            int idx = tid + i * 128;                   // 512 uint4 slots
            int row = idx >> 3, c8 = (idx & 7) * 8;
            *(uint4*)&Klds[row * KVSTRIDE + c8] =
                *(const uint4*)(Kp + (size_t)(kb + row) * HEADDIM + c8);
            uint4 raw = *(const uint4*)(Vp + (size_t)(kb + row) * HEADDIM + c8);
            const _Float16* hp = (const _Float16*)&raw;
#pragma unroll
            for (int t = 0; t < 8; ++t)
                Vt[(c8 + t) * KVSTRIDE + row] = hp[t];
        }
        __syncthreads();

        // ---- S = Q K^T (16x64 per wave, 8 WMMA) ----------------------------
        v8f s[4];
#pragma unroll
        for (int ni = 0; ni < 4; ++ni)
            s[ni] = (v8f){0.f, 0.f, 0.f, 0.f, 0.f, 0.f, 0.f, 0.f};
#pragma unroll
        for (int ni = 0; ni < 4; ++ni) {
            int krow = ni * 16 + lo;
#pragma unroll
            for (int kc = 0; kc < 2; ++kc) {
                Frag16 kbf;
#pragma unroll
                for (int v = 0; v < 8; ++v)
                    kbf.u[v] = *(const unsigned int*)
                        &Klds[krow * KVSTRIDE + kc * 32 + b_kidx(v, hf)];
                s[ni] = wmma_f16(qa[kc], kbf, s[ni]);
            }
        }

        // ---- scale + causal mask (diagonal tile only) ----------------------
        const bool diag = (j == qt);
#pragma unroll
        for (int ni = 0; ni < 4; ++ni)
#pragma unroll
            for (int r = 0; r < 8; ++r) {
                float v = s[ni][r] * scale;
                if (diag) {
                    int qg = wave * 16 + r + hf * 8;   // local query index
                    int kg = ni * 16 + lo;             // local key index
                    if (kg > qg) v = -1.0e30f;
                }
                s[ni][r] = v;
            }

        // ---- online softmax: row stats live in the 16-lane half owning row -
        float mnew[8], csc[8], rs[8];
#pragma unroll
        for (int r = 0; r < 8; ++r) {
            float t = s[0][r];
#pragma unroll
            for (int ni = 1; ni < 4; ++ni) t = fmaxf(t, s[ni][r]);
            t = fmaxf(t, __shfl_xor(t, 8, 32));
            t = fmaxf(t, __shfl_xor(t, 4, 32));
            t = fmaxf(t, __shfl_xor(t, 2, 32));
            t = fmaxf(t, __shfl_xor(t, 1, 32));
            mnew[r] = fmaxf(mprev[r], t);
            csc[r]  = __expf(mprev[r] - mnew[r]);
            rs[r]   = 0.f;
        }
        // p = exp(s - m), stash f16 P in per-wave LDS (C/D layout -> A layout)
#pragma unroll
        for (int ni = 0; ni < 4; ++ni)
#pragma unroll
            for (int r = 0; r < 8; ++r) {
                float p = __expf(s[ni][r] - mnew[r]);
                rs[r] += p;
                Plds[(wave * 16 + r + hf * 8) * KVSTRIDE + ni * 16 + lo] = (_Float16)p;
            }
#pragma unroll
        for (int r = 0; r < 8; ++r) {
            float t = rs[r];
            t += __shfl_xor(t, 8, 32);
            t += __shfl_xor(t, 4, 32);
            t += __shfl_xor(t, 2, 32);
            t += __shfl_xor(t, 1, 32);
            lsum[r]  = lsum[r] * csc[r] + t;
            mprev[r] = mnew[r];
        }
#pragma unroll
        for (int ni = 0; ni < 4; ++ni)
#pragma unroll
            for (int r = 0; r < 8; ++r)
                oacc[ni][r] *= csc[r];

        // ---- O += P V (8 WMMA) --------------------------------------------
#pragma unroll
        for (int kc = 0; kc < 2; ++kc) {
            Frag16 pf;
#pragma unroll
            for (int v = 0; v < 8; ++v)
                pf.u[v] = *(const unsigned int*)
                    &Plds[(wave * 16 + lo) * KVSTRIDE + kc * 32 + a_kidx(v, hf)];
#pragma unroll
            for (int ni = 0; ni < 4; ++ni) {
                Frag16 vb;
#pragma unroll
                for (int v = 0; v < 8; ++v)
                    vb.u[v] = *(const unsigned int*)
                        &Vt[(ni * 16 + lo) * KVSTRIDE + kc * 32 + b_kidx(v, hf)];
                oacc[ni] = wmma_f16(pf, vb, oacc[ni]);
            }
        }
    }

    // ---- normalize and write O in [B,S,H,HD] (= token-major [8192,1024]) --
    const int bb = bh >> 4, h = bh & 15;
#pragma unroll
    for (int ni = 0; ni < 4; ++ni)
#pragma unroll
        for (int r = 0; r < 8; ++r) {
            int qg = qbase + wave * 16 + r + hf * 8;
            int hd = ni * 16 + lo;
            float val = oacc[ni][r] / lsum[r];
            obuf[((size_t)(bb * SEQ + qg) * NHEADS + h) * HEADDIM + hd] = (_Float16)val;
        }
}

// ---------------------------------------------------------------------------
extern "C" void kernel_launch(void* const* d_in, const int* in_sizes, int n_in,
                              void* d_out, int out_size, void* d_ws, size_t ws_size,
                              hipStream_t stream) {
    const float* x     = (const float*)d_in[0];
    // d_in[1]: causal mask (bool) -- masking is done analytically, not read.
    const float* w_qkv = (const float*)d_in[2];
    const float* b_qkv = (const float*)d_in[3];
    const float* w_out = (const float*)d_in[4];
    const float* b_out = (const float*)d_in[5];
    float* out = (float*)d_out;

    const size_t HSZ = (size_t)BATCH * NHEADS * SEQ * HEADDIM;  // 8M halfs
    _Float16* ws = (_Float16*)d_ws;
    _Float16* qf = ws;
    _Float16* kf = ws + HSZ;
    _Float16* vf = ws + 2 * HSZ;
    _Float16* ob = ws + 3 * HSZ;

    // 1) QKV projection: [8192,1024] x [1024,3072] + bias -> f16 Q/K/V.
    gemm128<0><<<dim3(3 * DMODEL / 128, MTOT / 128), 256, 0, stream>>>(
        x, nullptr, w_qkv, b_qkv, qf, kf, vf, nullptr, DMODEL, 3 * DMODEL);

    // 2) Causal flash attention.
    flash_attn<<<dim3(SEQ / 64, BATCH * NHEADS), 128, 0, stream>>>(qf, kf, vf, ob);

    // 3) Output projection: [8192,1024] x [1024,1024] + bias -> f32 out.
    gemm128<1><<<dim3(DMODEL / 128, MTOT / 128), 256, 0, stream>>>(
        nullptr, ob, w_out, b_out, nullptr, nullptr, nullptr, out, DMODEL, DMODEL);
}
